// Transformer_E_55542517072407
// MI455X (gfx1250) — matrix operand order repaired and checked by reading, not compiled, not run on hardware
//
#include <hip/hip_runtime.h>

// ---------------- problem constants (match reference) ----------------
#define BB    2
#define SS    2048
#define DIMC  512
#define HC    8
#define DHC   64
#define MLPC  2048
#define CBSC  4
#define NBC   512          // SS / CBSC
#define MC    (BB*SS)      // 4096 rows
#define NEGC  (-1.0e9f)

typedef __attribute__((ext_vector_type(16))) _Float16 v16h;
typedef __attribute__((ext_vector_type(8)))  _Float16 v8h;
typedef __attribute__((ext_vector_type(8)))  float    v8f;

// ============================================================
// Weight convert + transpose f32 (K x N) -> f16 (N x K)
// ============================================================
__global__ void convT_kernel(const float* __restrict__ W, _Float16* __restrict__ Wt,
                             int K, int N) {
  int idx = blockIdx.x * 256 + threadIdx.x;
  if (idx >= K * N) return;
  int n = idx % N, k = idx / N;
  Wt[(size_t)n * K + k] = (_Float16)W[idx];
}

// ============================================================
// LayerNorm (+ optional fused RMSNorm).  One 256-thread block per row.
// ============================================================
template<bool RMS>
__global__ void ln_kernel(const float* __restrict__ x, const float* __restrict__ sc,
                          const float* __restrict__ bi, const float* __restrict__ g,
                          float* __restrict__ outf, _Float16* __restrict__ outh) {
  __shared__ float red[256];
  int row = blockIdx.x, t = threadIdx.x;
  const float* xp = x + (size_t)row * DIMC;
  float a0 = xp[t], a1 = xp[t + 256];
  red[t] = a0 + a1; __syncthreads();
  for (int s2 = 128; s2 > 0; s2 >>= 1) { if (t < s2) red[t] += red[t + s2]; __syncthreads(); }
  float mean = red[0] * (1.f / DIMC); __syncthreads();
  float d0 = a0 - mean, d1 = a1 - mean;
  red[t] = d0 * d0 + d1 * d1; __syncthreads();
  for (int s2 = 128; s2 > 0; s2 >>= 1) { if (t < s2) red[t] += red[t + s2]; __syncthreads(); }
  float inv = rsqrtf(red[0] * (1.f / DIMC) + 1e-5f); __syncthreads();
  float n0 = d0 * inv * sc[t]       + bi[t];
  float n1 = d1 * inv * sc[t + 256] + bi[t + 256];
  if (!RMS) {
    outh[(size_t)row * DIMC + t]       = (_Float16)n0;
    outh[(size_t)row * DIMC + t + 256] = (_Float16)n1;
    return;
  }
  red[t] = n0 * n0 + n1 * n1; __syncthreads();
  for (int s2 = 128; s2 > 0; s2 >>= 1) { if (t < s2) red[t] += red[t + s2]; __syncthreads(); }
  float rinv = rsqrtf(red[0] + 1e-12f) * 22.627416997969522f; // sqrt(512)
  float r0 = n0 * rinv * g[t], r1 = n1 * rinv * g[t + 256];
  outf[(size_t)row * DIMC + t]       = r0;
  outf[(size_t)row * DIMC + t + 256] = r1;
  outh[(size_t)row * DIMC + t]       = (_Float16)r0;
  outh[(size_t)row * DIMC + t + 256] = (_Float16)r1;
}

// ============================================================
// LDS-tiled WMMA GEMM with async global->LDS staging (CDNA5 ASYNCcnt path).
//   C(MxN) = A(MxK,f16) * Bt(NxK,f16)^T  (+ epilogue)
// Block tile 128x64, 8 waves (4x2), each wave 32x32 = 2x2 wmma accums.
// BK=64 staged to LDS with GLOBAL_LOAD_ASYNC_TO_LDS_B128, double buffered.
// EPI: 0 = store f32; 1 = +residual f32; 2 = +bias,leaky->f16; 3 = +bias+residual f32
// ============================================================
#define BMT 128
#define BNT 64
#define BKT 64
#define APADH 72
#define A_LDS_H (BMT * APADH)
#define B_LDS_H (BNT * APADH)
#define BUF_H   (A_LDS_H + B_LDS_H)

template<int EPI>
__global__ __launch_bounds__(256)
void gemm_lds_wmma(const _Float16* __restrict__ A, const _Float16* __restrict__ Bt,
                   const float* __restrict__ bias, const float* __restrict__ R,
                   float* __restrict__ Cf, _Float16* __restrict__ Ch,
                   int M, int N, int K) {
  __shared__ __align__(32) _Float16 smem[2 * BUF_H];
  const int tid  = threadIdx.x;
  const int lane = tid & 31;
  const int wv   = tid >> 5;
  const int wm   = wv & 3;
  const int wn   = wv >> 2;
  const int nBlocksN = N / BNT;
  const int m0 = (blockIdx.x / nBlocksN) * BMT;
  const int n0 = (blockIdx.x % nBlocksN) * BNT;
  const uint32_t lds0 = (uint32_t)(uintptr_t)(&smem[0]);

  auto stage = [&](int buf, int k0) {
    uint32_t base = lds0 + (uint32_t)buf * (BUF_H * 2);
#pragma unroll
    for (int c = 0; c < 4; ++c) {
      int cidx = tid + c * 256;
      int r  = cidx >> 3;
      int cc = cidx & 7;
      uint32_t loff = base + (uint32_t)r * (APADH * 2) + (uint32_t)cc * 16;
      uint64_t g = (uint64_t)(uintptr_t)(A + (size_t)(m0 + r) * K + k0 + cc * 8);
      asm volatile("global_load_async_to_lds_b128 %0, %1, off"
                   :: "v"(loff), "v"(g) : "memory");
    }
#pragma unroll
    for (int c = 0; c < 2; ++c) {
      int cidx = tid + c * 256;
      int r  = cidx >> 3;
      int cc = cidx & 7;
      uint32_t loff = base + (uint32_t)(A_LDS_H * 2) + (uint32_t)r * (APADH * 2) + (uint32_t)cc * 16;
      uint64_t g = (uint64_t)(uintptr_t)(Bt + (size_t)(n0 + r) * K + k0 + cc * 8);
      asm volatile("global_load_async_to_lds_b128 %0, %1, off"
                   :: "v"(loff), "v"(g) : "memory");
    }
  };

  v8f acc[2][2] = {};
  const int niter = K / BKT;
  stage(0, 0);
  const int row16 = lane & 15, khalf = lane >> 4;
  for (int it = 0; it < niter; ++it) {
    asm volatile("s_wait_asynccnt 0x0" ::: "memory");
    __syncthreads();
    if (it + 1 < niter) stage((it + 1) & 1, (it + 1) * BKT);
    const _Float16* abuf = smem + (size_t)(it & 1) * BUF_H;
    const _Float16* bbuf = abuf + A_LDS_H;
#pragma unroll
    for (int kk = 0; kk < 2; ++kk) {
      v16h afr[2], bfr[2];
#pragma unroll
      for (int ms = 0; ms < 2; ++ms) {
        const _Float16* ap = abuf + (size_t)(wm * 32 + ms * 16 + row16) * APADH + kk * 32 + khalf * 8;
        v8h lo = *(const v8h*)ap;
        v8h hi = *(const v8h*)(ap + 16);
#pragma unroll
        for (int e = 0; e < 8; ++e) { afr[ms][e] = lo[e]; afr[ms][8 + e] = hi[e]; }
      }
#pragma unroll
      for (int ns = 0; ns < 2; ++ns) {
        const _Float16* bp = bbuf + (size_t)(wn * 32 + ns * 16 + row16) * APADH + kk * 32 + khalf * 8;
        v8h lo = *(const v8h*)bp;
        v8h hi = *(const v8h*)(bp + 16);
#pragma unroll
        for (int e = 0; e < 8; ++e) { bfr[ns][e] = lo[e]; bfr[ns][8 + e] = hi[e]; }
      }
#pragma unroll
      for (int ms = 0; ms < 2; ++ms)
#pragma unroll
        for (int ns = 0; ns < 2; ++ns)
          acc[ms][ns] = __builtin_amdgcn_wmma_f32_16x16x32_f16(
              false, afr[ms], false, bfr[ns], (short)0, acc[ms][ns], false, false);
    }
    __syncthreads();
  }
  const int cnb = n0 + wn * 32 + (lane & 15);
  const int mbb = m0 + wm * 32 + (lane >> 4) * 8;
#pragma unroll
  for (int ms = 0; ms < 2; ++ms) {
#pragma unroll
    for (int ns = 0; ns < 2; ++ns) {
      int cn = cnb + ns * 16;
      float bv = (EPI == 2 || EPI == 3) ? bias[cn] : 0.f;
#pragma unroll
      for (int r = 0; r < 8; ++r) {
        size_t off = (size_t)(mbb + ms * 16 + r) * N + cn;
        float val = acc[ms][ns][r] + bv;
        if (EPI == 2) {
          val = val > 0.f ? val : 0.01f * val;
          Ch[off] = (_Float16)val;
        } else {
          if (EPI == 1 || EPI == 3) val += R[off];
          Cf[off] = val;
        }
      }
    }
  }
}

// ============================================================
// RoPE in place on q and k; also emits f16 copy of roped q for WMMA attention.
// ============================================================
__global__ void rope_kernel(float* __restrict__ q, float* __restrict__ k,
                            _Float16* __restrict__ qh) {
  int idx = blockIdx.x * 256 + threadIdx.x;
  if (idx >= MC * HC * 32) return;
  int j = idx & 31;
  int h = (idx >> 5) & (HC - 1);
  int m = idx >> 8;
  int s = m & (SS - 1);
  float freq = __expf(-(float)j * (9.210340371976184f / 32.f)); // 10000^(-j/32)
  float ang = (float)s * freq;
  float cs = cosf(ang), sn = sinf(ang);
  size_t base = (size_t)m * DIMC + h * DHC + j;
  float q1 = q[base], q2 = q[base + 32];
  float qa = q1 * cs - q2 * sn, qb2 = q1 * sn + q2 * cs;
  q[base] = qa; q[base + 32] = qb2;
  qh[base] = (_Float16)qa; qh[base + 32] = (_Float16)qb2;
  float k1 = k[base], k2 = k[base + 32];
  k[base]      = k1 * cs - k2 * sn;
  k[base + 32] = k1 * sn + k2 * cs;
}

// ============================================================
// ck/cv build: out (b,h, 513, 64); slot 0 = mem vector
// ============================================================
__global__ void ckcv_kernel(const float* __restrict__ kv, const float* __restrict__ pos,
                            const float* __restrict__ Wc, const float* __restrict__ memv,
                            float* __restrict__ out) {
  int idx = blockIdx.x * 256 + threadIdx.x;
  if (idx >= BB * HC * (NBC + 1) * DHC) return;
  int d = idx & 63;
  int r = idx >> 6;
  int n = r % (NBC + 1);
  int h = (r / (NBC + 1)) % HC;
  int b = r / ((NBC + 1) * HC);
  if (n == 0) { out[idx] = memv[h * DHC + d]; return; }
  float acc = 0.f;
  int tok0 = (n - 1) * CBSC;
  for (int t2 = 0; t2 < CBSC; ++t2) {
    const float* kp = kv  + ((size_t)(b * SS + tok0 + t2)) * DIMC + h * DHC;
    const float* pp = pos + (h * CBSC + t2) * DHC;
    const float* wp = Wc  + ((size_t)(h * CBSC * DHC + t2 * DHC)) * DHC + d;
    for (int e = 0; e < DHC; ++e) acc += (kp[e] + pp[e]) * wp[(size_t)e * DHC];
  }
  out[idx] = acc;
}

// ============================================================
// Build f16 padded CK (bh,544,64) and transposed CV^T (bh,64,544) for WMMA attention.
// ============================================================
#define NPAD 544            // 513 padded to 34 tiles of 16 (= 17 wmma K-steps)
#define NT34 (NPAD / 16)
#define SROW 548            // scs row stride (floats): 8-row offset not bank-aligned
#define PROW 552            // p16 row stride (halves): *2 multiple of 16B

__global__ void ckcvh_kernel(const float* __restrict__ ckb, const float* __restrict__ cvb,
                             _Float16* __restrict__ ckh, _Float16* __restrict__ cvth) {
  int idx = blockIdx.x * 256 + threadIdx.x;
  if (idx >= BB * HC * NPAD * DHC) return;
  int d = idx & 63;
  int r = idx >> 6;
  int j = r % NPAD;
  int bh = r / NPAD;
  float kv = 0.f, vv = 0.f;
  if (j < NBC + 1) {
    size_t src = ((size_t)bh * (NBC + 1) + j) * DHC + d;
    kv = ckb[src]; vv = cvb[src];
  }
  ckh[((size_t)bh * NPAD + j) * DHC + d]  = (_Float16)kv;
  cvth[((size_t)bh * DHC + d) * NPAD + j] = (_Float16)vv;
}

// ============================================================
// WMMA compressed attention + fused softmax + top-2 block selection.
// One block per (b,h, 32 queries); 256 threads = 8 waves.
//  Phase1: S = Q*CK^T (2 Mtiles x 34 Ntiles, K=64) -> masked/scaled scores in LDS
//  Phase2: per-row softmax stats (8 lanes/row, shfl) -> unnormalized exp as f16 P in LDS,
//          fused top-2 selection over blocks (exclude own)
//  Phase3: OUT = P*CV^T via WMMA (8 tiles of 16x16, 17 K-steps), scaled by 1/sum
// ============================================================
#define QBQ 32

__global__ __launch_bounds__(256)
void cattn_wmma_kernel(const _Float16* __restrict__ qh, const _Float16* __restrict__ ckh,
                       const _Float16* __restrict__ cvth, float* __restrict__ out_c,
                       int* __restrict__ sel) {
  extern __shared__ char dynsmem[];
  float*    scs    = (float*)dynsmem;                                   // QBQ*SROW f32
  _Float16* p16    = (_Float16*)(dynsmem + QBQ * SROW * 4);             // QBQ*PROW f16
  float*    rowinv = (float*)(dynsmem + QBQ * SROW * 4 + QBQ * PROW * 2);

  const int nQB = SS / QBQ;                 // 64
  const int bh   = blockIdx.x / nQB;        // (b*H + h)
  const int qblk = blockIdx.x % nQB;
  const int h = bh & (HC - 1);
  const int b = bh >> 3;
  const int i0 = qblk * QBQ;
  const int tid = threadIdx.x, lane = tid & 31, wv = tid >> 5;
  const int row16 = lane & 15, khalf = lane >> 4;

  // ---- Phase 1: scores ----
  {
    const int mt = wv & 1;                  // query sub-tile (16)
    const int ng = wv >> 1;                 // n-tile group 0..3
    v16h afr[2];
    const _Float16* qrowp = qh + ((size_t)(b * SS + i0 + mt * 16 + row16)) * DIMC + h * DHC;
#pragma unroll
    for (int kk = 0; kk < 2; ++kk) {
      v8h lo = *(const v8h*)(qrowp + kk * 32 + khalf * 8);
      v8h hi = *(const v8h*)(qrowp + kk * 32 + khalf * 8 + 16);
#pragma unroll
      for (int e = 0; e < 8; ++e) { afr[kk][e] = lo[e]; afr[kk][8 + e] = hi[e]; }
    }
    const _Float16* ckbh = ckh + (size_t)bh * NPAD * DHC;
    for (int jt = ng; jt < NT34; jt += 4) {
      v8f acc = {};
      const _Float16* ckr = ckbh + (size_t)(jt * 16 + row16) * DHC;
#pragma unroll
      for (int kk = 0; kk < 2; ++kk) {
        v8h lo = *(const v8h*)(ckr + kk * 32 + khalf * 8);
        v8h hi = *(const v8h*)(ckr + kk * 32 + khalf * 8 + 16);
        v16h bfr;
#pragma unroll
        for (int e = 0; e < 8; ++e) { bfr[e] = lo[e]; bfr[8 + e] = hi[e]; }
        acc = __builtin_amdgcn_wmma_f32_16x16x32_f16(false, afr[kk], false, bfr,
                                                     (short)0, acc, false, false);
      }
      int j   = jt * 16 + (lane & 15);
      int qr0 = mt * 16 + (lane >> 4) * 8;
#pragma unroll
      for (int r = 0; r < 8; ++r) {
        int i = i0 + qr0 + r;
        bool ok = (j <= NBC) && ((j == 0) || (i >= j * CBSC - 1));  // pos >= blk_end
        scs[(qr0 + r) * SROW + j] = ok ? acc[r] * 0.125f : NEGC;
      }
    }
  }
  __syncthreads();

  // ---- Phase 2: softmax stats + f16 P + top-2 ----
  {
    const int qrow = tid >> 3, l8 = tid & 7;
    const float* srow = scs + (size_t)qrow * SROW;
    float mx = -3.4e38f;
    for (int j = l8; j <= NBC; j += 8) mx = fmaxf(mx, srow[j]);
#pragma unroll
    for (int m2 = 1; m2 < 8; m2 <<= 1) mx = fmaxf(mx, __shfl_xor(mx, m2, 8));
    float sum = 0.f;
    _Float16* prow = p16 + (size_t)qrow * PROW;
    const int own = (i0 + qrow) >> 2;
    float mv1 = -1.f, mv2 = -1.f; int mi1 = 1 << 29, mi2 = 1 << 29;
    for (int j = l8; j < NPAD; j += 8) {
      float e = (j <= NBC) ? __expf(srow[j] - mx) : 0.f;
      prow[j] = (_Float16)e;
      sum += e;
      int jb = j - 1;
      if (j >= 1 && j <= NBC && jb != own) {
        if (e > mv1 || (e == mv1 && jb < mi1)) { mv2 = mv1; mi2 = mi1; mv1 = e; mi1 = jb; }
        else if (e > mv2 || (e == mv2 && jb < mi2)) { mv2 = e; mi2 = jb; }
      }
    }
#pragma unroll
    for (int m2 = 1; m2 < 8; m2 <<= 1) sum += __shfl_xor(sum, m2, 8);
#pragma unroll
    for (int m2 = 1; m2 < 8; m2 <<= 1) {
      float w1 = __shfl_xor(mv1, m2, 8); int j1 = __shfl_xor(mi1, m2, 8);
      float w2 = __shfl_xor(mv2, m2, 8); int j2 = __shfl_xor(mi2, m2, 8);
      if (w1 > mv1 || (w1 == mv1 && j1 < mi1)) { mv2 = mv1; mi2 = mi1; mv1 = w1; mi1 = j1; }
      else if (w1 > mv2 || (w1 == mv2 && j1 < mi2)) { mv2 = w1; mi2 = j1; }
      if (w2 > mv1 || (w2 == mv1 && j2 < mi1)) { mv2 = mv1; mi2 = mi1; mv1 = w2; mi1 = j2; }
      else if (w2 > mv2 || (w2 == mv2 && j2 < mi2)) { mv2 = w2; mi2 = j2; }
    }
    if (l8 == 0) {
      rowinv[qrow] = 1.f / sum;
      size_t qi = (size_t)bh * SS + i0 + qrow;
      sel[qi * 3] = mi1; sel[qi * 3 + 1] = mi2; sel[qi * 3 + 2] = own;
    }
  }
  __syncthreads();

  // ---- Phase 3: OUT = (P/sum) * CV ----
  {
    const int mt = wv & 1;                  // query sub-tile
    const int nt = wv >> 1;                 // output-dim tile 0..3
    v8f acc = {};
    const _Float16* prow0 = p16 + (size_t)(mt * 16 + row16) * PROW;
    const _Float16* cvr = cvth + ((size_t)bh * DHC + nt * 16 + row16) * NPAD;
    for (int k0 = 0; k0 < NPAD; k0 += 32) {
      v16h afr, bfr;
      v8h alo = *(const v8h*)(prow0 + k0 + khalf * 8);
      v8h ahi = *(const v8h*)(prow0 + k0 + khalf * 8 + 16);
      v8h blo = *(const v8h*)(cvr + k0 + khalf * 8);
      v8h bhi = *(const v8h*)(cvr + k0 + khalf * 8 + 16);
#pragma unroll
      for (int e = 0; e < 8; ++e) { afr[e] = alo[e]; afr[8 + e] = ahi[e]; bfr[e] = blo[e]; bfr[8 + e] = bhi[e]; }
      acc = __builtin_amdgcn_wmma_f32_16x16x32_f16(false, afr, false, bfr, (short)0, acc, false, false);
    }
    int d   = nt * 16 + (lane & 15);
    int qr0 = mt * 16 + (lane >> 4) * 8;
#pragma unroll
    for (int r = 0; r < 8; ++r) {
      int qrow = qr0 + r;
      size_t qi = (size_t)bh * SS + i0 + qrow;
      out_c[qi * DHC + d] = acc[r] * rowinv[qrow];
    }
  }
}

// ============================================================
// Fine (3 blocks x 4 tokens) + sliding window (2) + sigmoid gates, combine.
// ============================================================
__global__ void fine_combine_kernel(const float* __restrict__ q, const float* __restrict__ k,
                                    const float* __restrict__ v, const float* __restrict__ xr,
                                    const float* __restrict__ Wgl, const float* __restrict__ out_c,
                                    const int* __restrict__ sel, _Float16* __restrict__ attnh) {
  int qi = blockIdx.x;
  int i = qi & (SS - 1);
  int h = (qi >> 11) & (HC - 1);
  int b = qi >> 14;
  __shared__ float qs[DHC];
  __shared__ float ps[14];
  __shared__ int   toks[14];
  __shared__ float gg[3];
  int t = threadIdx.x;
  qs[t] = q[((size_t)(b * SS + i)) * DIMC + h * DHC + t];
  __syncthreads();
  const float scale = 0.125f;
  if (t < 12) {
    int blk = sel[qi * 3 + (t >> 2)];
    int tok = blk * CBSC + (t & 3);
    int tc = tok < SS ? tok : SS - 1;
    toks[t] = tc;
    const float* kp = k + ((size_t)(b * SS + tc)) * DIMC + h * DHC;
    float dot = 0.f; for (int d = 0; d < DHC; ++d) dot += qs[d] * kp[d];
    ps[t] = (tok <= i) ? dot * scale : NEGC;
  } else if (t < 14) {
    int w = t - 12; int tok = i - w; int tc = tok < 0 ? 0 : tok;
    toks[t] = tc;
    const float* kp = k + ((size_t)(b * SS + tc)) * DIMC + h * DHC;
    float dot = 0.f; for (int d = 0; d < DHC; ++d) dot += qs[d] * kp[d];
    ps[t] = (tok >= 0) ? dot * scale : NEGC;
  } else if (t < 17) {
    int gk = t - 14;
    const float* xp = xr + ((size_t)(b * SS + i)) * DIMC;
    float dot = 0.f;
    for (int c = 0; c < DIMC; ++c) dot += xp[c] * Wgl[(size_t)c * (3 * HC) + h * 3 + gk];
    gg[gk] = 1.f / (1.f + __expf(-dot));
  }
  __syncthreads();
  if (t == 0) {
    float mxf = -3.4e38f;
    for (int j2 = 0; j2 < 12; ++j2) mxf = fmaxf(mxf, ps[j2]);
    float sm = 0.f;
    for (int j2 = 0; j2 < 12; ++j2) { float e = __expf(ps[j2] - mxf); ps[j2] = e; sm += e; }
    float invv = 1.f / sm;
    for (int j2 = 0; j2 < 12; ++j2) ps[j2] *= invv;
    float m2 = fmaxf(ps[12], ps[13]);
    float e0 = __expf(ps[12] - m2), e1 = __expf(ps[13] - m2);
    float ss2 = e0 + e1; ps[12] = e0 / ss2; ps[13] = e1 / ss2;
  }
  __syncthreads();
  float of = 0.f, os = 0.f;
  for (int j2 = 0; j2 < 12; ++j2)
    of += ps[j2] * v[((size_t)(b * SS + toks[j2])) * DIMC + h * DHC + t];
  for (int j2 = 12; j2 < 14; ++j2)
    os += ps[j2] * v[((size_t)(b * SS + toks[j2])) * DIMC + h * DHC + t];
  float o = gg[0] * out_c[(size_t)qi * DHC + t] + gg[1] * of + gg[2] * os;
  attnh[((size_t)(b * SS + i)) * DIMC + h * DHC + t] = (_Float16)o;
}

// ============================================================
// Host orchestration
// ============================================================
extern "C" void kernel_launch(void* const* d_in, const int* in_sizes, int n_in,
                              void* d_out, int out_size, void* d_ws, size_t ws_size,
                              hipStream_t stream) {
  (void)in_sizes; (void)n_in; (void)out_size; (void)ws_size;
  const float* x_in  = (const float*)d_in[0];
  const float* ln1_s = (const float*)d_in[1];
  const float* ln1_b = (const float*)d_in[2];
  const float* rms_g = (const float*)d_in[3];
  const float* Wq    = (const float*)d_in[4];
  const float* Wk    = (const float*)d_in[5];
  const float* Wv    = (const float*)d_in[6];
  const float* kpos  = (const float*)d_in[7];
  const float* vpos  = (const float*)d_in[8];
  const float* Wck   = (const float*)d_in[9];
  const float* Wcv   = (const float*)d_in[10];
  const float* mem_k = (const float*)d_in[11];
  const float* mem_v = (const float*)d_in[12];
  const float* Wg    = (const float*)d_in[13];
  const float* Wo    = (const float*)d_in[14];
  const float* ln2_s = (const float*)d_in[15];
  const float* ln2_b = (const float*)d_in[16];
  const float* W1    = (const float*)d_in[17];
  const float* b1    = (const float*)d_in[18];
  const float* W2    = (const float*)d_in[19];
  const float* b2    = (const float*)d_in[20];

  char* ws = (char*)d_ws;
  size_t off = 0;
  auto alloc = [&](size_t bytes) -> char* {
    off = (off + 255) & ~(size_t)255;
    char* p = ws + off; off += bytes; return p;
  };
  float*     xbuf  = (float*)    alloc((size_t)MC * DIMC * 4);
  float*     xr    = (float*)    alloc((size_t)MC * DIMC * 4);
  _Float16*  xrh   = (_Float16*) alloc((size_t)MC * DIMC * 2);
  float*     qb    = (float*)    alloc((size_t)MC * DIMC * 4);
  float*     kb    = (float*)    alloc((size_t)MC * DIMC * 4);
  float*     vb    = (float*)    alloc((size_t)MC * DIMC * 4);
  _Float16*  qhb   = (_Float16*) alloc((size_t)MC * DIMC * 2);
  float*     ckb   = (float*)    alloc((size_t)BB * HC * (NBC + 1) * DHC * 4);
  float*     cvb   = (float*)    alloc((size_t)BB * HC * (NBC + 1) * DHC * 4);
  _Float16*  ckh   = (_Float16*) alloc((size_t)BB * HC * NPAD * DHC * 2);
  _Float16*  cvth  = (_Float16*) alloc((size_t)BB * HC * DHC * NPAD * 2);
  float*     ocb   = (float*)    alloc((size_t)BB * HC * SS * DHC * 4);
  int*       selb  = (int*)      alloc((size_t)BB * HC * SS * 3 * 4);
  _Float16*  attnh = (_Float16*) alloc((size_t)MC * DIMC * 2);
  _Float16*  hh    = (_Float16*) alloc((size_t)MC * DIMC * 2);
  _Float16*  h1h   = (_Float16*) alloc((size_t)MC * MLPC * 2);
  _Float16*  wqt   = (_Float16*) alloc((size_t)DIMC * DIMC * 2);
  _Float16*  wkt   = (_Float16*) alloc((size_t)DIMC * DIMC * 2);
  _Float16*  wvt   = (_Float16*) alloc((size_t)DIMC * DIMC * 2);
  _Float16*  wot   = (_Float16*) alloc((size_t)DIMC * DIMC * 2);
  _Float16*  w1t   = (_Float16*) alloc((size_t)DIMC * MLPC * 2);
  _Float16*  w2t   = (_Float16*) alloc((size_t)MLPC * DIMC * 2);

  hipMemcpyAsync(xbuf, x_in, (size_t)MC * DIMC * 4, hipMemcpyDeviceToDevice, stream);

  const int gb512  = (MC / BMT) * (DIMC / BNT);  // 256 blocks
  const int gb2048 = (MC / BMT) * (MLPC / BNT);  // 1024 blocks
  const int nQ = BB * HC * SS;                   // 32768 queries
  const int cattnLds = QBQ * SROW * 4 + QBQ * PROW * 2 + QBQ * 4;  // ~105.7 KB

  for (int l = 0; l < 2; ++l) {
    const float* Wq_l  = Wq + (size_t)l * DIMC * DIMC;
    const float* Wk_l  = Wk + (size_t)l * DIMC * DIMC;
    const float* Wv_l  = Wv + (size_t)l * DIMC * DIMC;
    const float* Wo_l  = Wo + (size_t)l * DIMC * DIMC;
    const float* W1_l  = W1 + (size_t)l * DIMC * MLPC;
    const float* W2_l  = W2 + (size_t)l * MLPC * DIMC;

    convT_kernel<<<(DIMC*DIMC + 255)/256, 256, 0, stream>>>(Wq_l, wqt, DIMC, DIMC);
    convT_kernel<<<(DIMC*DIMC + 255)/256, 256, 0, stream>>>(Wk_l, wkt, DIMC, DIMC);
    convT_kernel<<<(DIMC*DIMC + 255)/256, 256, 0, stream>>>(Wv_l, wvt, DIMC, DIMC);
    convT_kernel<<<(DIMC*DIMC + 255)/256, 256, 0, stream>>>(Wo_l, wot, DIMC, DIMC);
    convT_kernel<<<(DIMC*MLPC + 255)/256, 256, 0, stream>>>(W1_l, w1t, DIMC, MLPC);
    convT_kernel<<<(MLPC*DIMC + 255)/256, 256, 0, stream>>>(W2_l, w2t, MLPC, DIMC);

    ln_kernel<true><<<MC, 256, 0, stream>>>(xbuf, ln1_s + l*DIMC, ln1_b + l*DIMC,
                                            rms_g + l*DIMC, xr, xrh);

    gemm_lds_wmma<0><<<gb512, 256, 0, stream>>>(xrh, wqt, nullptr, nullptr, qb, nullptr, MC, DIMC, DIMC);
    gemm_lds_wmma<0><<<gb512, 256, 0, stream>>>(xrh, wkt, nullptr, nullptr, kb, nullptr, MC, DIMC, DIMC);
    gemm_lds_wmma<0><<<gb512, 256, 0, stream>>>(xrh, wvt, nullptr, nullptr, vb, nullptr, MC, DIMC, DIMC);

    rope_kernel<<<(MC*HC*32)/256, 256, 0, stream>>>(qb, kb, qhb);

    int ckn = BB * HC * (NBC + 1) * DHC;
    ckcv_kernel<<<(ckn + 255)/256, 256, 0, stream>>>(
        kb, kpos + (size_t)l*HC*CBSC*DHC, Wck + (size_t)l*HC*CBSC*DHC*DHC,
        mem_k + (size_t)l*HC*DHC, ckb);
    ckcv_kernel<<<(ckn + 255)/256, 256, 0, stream>>>(
        vb, vpos + (size_t)l*HC*CBSC*DHC, Wcv + (size_t)l*HC*CBSC*DHC*DHC,
        mem_v + (size_t)l*HC*DHC, cvb);

    int pkn = BB * HC * NPAD * DHC;
    ckcvh_kernel<<<(pkn + 255)/256, 256, 0, stream>>>(ckb, cvb, ckh, cvth);

    cattn_wmma_kernel<<<BB * HC * (SS / QBQ), 256, cattnLds, stream>>>(qhb, ckh, cvth, ocb, selb);

    fine_combine_kernel<<<nQ, 64, 0, stream>>>(qb, kb, vb, xr,
                                               Wg + (size_t)l*DIMC*3*HC, ocb, selb, attnh);

    gemm_lds_wmma<1><<<gb512, 256, 0, stream>>>(attnh, wot, nullptr, xbuf, xbuf, nullptr, MC, DIMC, DIMC);

    ln_kernel<false><<<MC, 256, 0, stream>>>(xbuf, ln2_s + l*DIMC, ln2_b + l*DIMC,
                                             nullptr, nullptr, hh);
    gemm_lds_wmma<2><<<gb2048, 256, 0, stream>>>(hh, w1t, b1 + (size_t)l*MLPC,
                                                 nullptr, nullptr, h1h, MC, MLPC, DIMC);
    float* outp = (l == 1) ? (float*)d_out : xbuf;
    gemm_lds_wmma<3><<<gb512, 256, 0, stream>>>(h1h, w2t, b2 + (size_t)l*DIMC,
                                                xbuf, outp, nullptr, MC, DIMC, MLPC);
  }
}
